// AttentionLayer_21517786153158
// MI455X (gfx1250) — matrix operand order repaired and checked by reading, not compiled, not run on hardware
//
#include <hip/hip_runtime.h>
#include <hip/hip_bf16.h>
#include <math.h>
#include <stdint.h>

typedef __attribute__((ext_vector_type(16))) __bf16 v16bf;
typedef __attribute__((ext_vector_type(8)))  __bf16 v8bf;
typedef __attribute__((ext_vector_type(8)))  float  v8f;
typedef __attribute__((ext_vector_type(4)))  int    v4i;

#define H_    4096
#define NH_   32
#define NG_   8
#define HD_   128
#define B_    2
#define S_    2048
#define NQKV  6144
#define MROWS (B_*S_)   // 4096

__device__ __forceinline__ __bf16 f2bf(float f) {
  unsigned u = __builtin_bit_cast(unsigned, f);
  unsigned r = (u + 0x7FFFu + ((u >> 16) & 1u)) >> 16;
  unsigned short s = (unsigned short)r;
  return __builtin_bit_cast(__bf16, s);
}
__device__ __forceinline__ float bf2f(__bf16 b) {
  unsigned short s = __builtin_bit_cast(unsigned short, b);
  unsigned u = ((unsigned)s) << 16;
  return __builtin_bit_cast(float, u);
}
__device__ __forceinline__ v16bf pack16(v8bf lo, v8bf hi) {
  v16bf r;
#pragma unroll
  for (int i = 0; i < 8; ++i) { r[i] = lo[i]; r[i + 8] = hi[i]; }
  return r;
}
__device__ __forceinline__ v8f zero8() {
  v8f z;
#pragma unroll
  for (int i = 0; i < 8; ++i) z[i] = 0.f;
  return z;
}

// --- CDNA5 async global->LDS copy (ASYNCcnt-tracked), 16B per lane ---------
__device__ __forceinline__ void async_copy_b128(const __bf16* g, __bf16* l) {
#if __has_builtin(__builtin_amdgcn_global_load_async_to_lds_b128)
  typedef __attribute__((address_space(1))) v4i* gp_t;
  typedef __attribute__((address_space(3))) v4i* lp_t;
  __builtin_amdgcn_global_load_async_to_lds_b128(
      (gp_t)(unsigned long long)(uintptr_t)g,
      (lp_t)(unsigned int)(uintptr_t)l, 0, 0);
#else
  unsigned int lo = (unsigned int)(uintptr_t)l;
  unsigned long long ga = (unsigned long long)(uintptr_t)g;
  asm volatile("global_load_async_to_lds_b128 %0, %1, off"
               :: "v"(lo), "v"(ga) : "memory");
#endif
}
__device__ __forceinline__ void wait_async0() {
#if __has_builtin(__builtin_amdgcn_s_wait_asynccnt)
  __builtin_amdgcn_s_wait_asynccnt(0);
#else
  asm volatile("s_wait_asynccnt 0x0" ::: "memory");
#endif
}

// ---------------------------------------------------------------------------
// Kernel 0a: elementwise f32 -> bf16
// ---------------------------------------------------------------------------
__global__ __launch_bounds__(256) void cvt_bf16_kernel(
    const float* __restrict__ in, __bf16* __restrict__ out, int n) {
  int i = blockIdx.x * 256 + threadIdx.x;
  if (i < n) out[i] = f2bf(in[i]);
}

// ---------------------------------------------------------------------------
// Kernel 0b: tiled transpose + convert: out[n*K + k] = bf16(in[k*N + n])
// ---------------------------------------------------------------------------
__global__ __launch_bounds__(256) void transpose_cvt_kernel(
    const float* __restrict__ in, __bf16* __restrict__ out, int K, int N) {
  __shared__ float tile[32][33];
  int bk = blockIdx.x * 32;
  int bn = blockIdx.y * 32;
  int tx = threadIdx.x & 31, ty = threadIdx.x >> 5;  // 32 x 8
#pragma unroll
  for (int j = 0; j < 32; j += 8)
    tile[ty + j][tx] = in[(size_t)(bk + ty + j) * N + bn + tx];
  __syncthreads();
#pragma unroll
  for (int j = 0; j < 32; j += 8)
    out[(size_t)(bn + ty + j) * K + bk + tx] = f2bf(tile[tx][ty + j]);
}

// ---------------------------------------------------------------------------
// GEMM: C[M,N] = A(bf16 [M,K]) @ Bt(bf16 [N,K])^T (+bias)
// 128x128 block tile, BK=32, 8 waves x (64x32 = 4x2 wmma tiles).
// Double-buffered LDS staging via async global->LDS b128 copies.
// ---------------------------------------------------------------------------
template <bool BF16_OUT>
__global__ __launch_bounds__(256) void gemm_kernel(
    const __bf16* __restrict__ A, const __bf16* __restrict__ Bt,
    const float* __restrict__ bias, void* __restrict__ Cv, const int N) {
  const int K = H_;
  __shared__ __bf16 As[2][128 * 40];  // [m][k], padded stride 40
  __shared__ __bf16 Bs[2][128 * 40];  // [n][k]
  const int tid = threadIdx.x;
  const int lane = tid & 31, wave = tid >> 5;
  const int half = lane >> 4, l16 = lane & 15;
  const int bm = blockIdx.y * 128, bn = blockIdx.x * 128;
  const int wm = (wave & 1) * 64, wn = (wave >> 1) * 32;

  auto stage = [&](int buf, int k0) {
#pragma unroll
    for (int j = 0; j < 2; ++j) {
      int c = tid + 256 * j;
      int r = c >> 2, q = (c & 3) * 8;
      async_copy_b128(A + (size_t)(bm + r) * K + k0 + q, &As[buf][r * 40 + q]);
      async_copy_b128(Bt + (size_t)(bn + r) * K + k0 + q, &Bs[buf][r * 40 + q]);
    }
  };

  v8f acc[4][2];
#pragma unroll
  for (int mi = 0; mi < 4; ++mi)
#pragma unroll
    for (int ni = 0; ni < 2; ++ni) acc[mi][ni] = zero8();

  stage(0, 0);
  const int NIT = K / 32;
  for (int it = 0; it < NIT; ++it) {
    wait_async0();       // my async writes to buf (it&1) done
    __syncthreads();     // all waves' writes landed; all done reading buf^1
    if (it + 1 < NIT) stage((it + 1) & 1, (it + 1) * 32);
    const __bf16* Ab = As[it & 1];
    const __bf16* Bb = Bs[it & 1];
    v16bf af[4], bfr[2];
#pragma unroll
    for (int mi = 0; mi < 4; ++mi) {
      const __bf16* p = &Ab[(wm + mi * 16 + l16) * 40 + half * 8];
      af[mi] = pack16(*(const v8bf*)p, *(const v8bf*)(p + 16));
    }
#pragma unroll
    for (int ni = 0; ni < 2; ++ni) {
      const __bf16* p = &Bb[(wn + ni * 16 + l16) * 40 + half * 8];
      bfr[ni] = pack16(*(const v8bf*)p, *(const v8bf*)(p + 16));
    }
#pragma unroll
    for (int mi = 0; mi < 4; ++mi)
#pragma unroll
      for (int ni = 0; ni < 2; ++ni)
        acc[mi][ni] = __builtin_amdgcn_wmma_f32_16x16x32_bf16(
            false, af[mi], false, bfr[ni], (short)0, acc[mi][ni], false, false);
  }
#pragma unroll
  for (int mi = 0; mi < 4; ++mi)
#pragma unroll
    for (int ni = 0; ni < 2; ++ni) {
      int col = bn + wn + ni * 16 + l16;
      float bv = BF16_OUT ? bias[col] : 0.f;
#pragma unroll
      for (int r = 0; r < 8; ++r) {
        int row = bm + wm + mi * 16 + r + half * 8;
        if constexpr (BF16_OUT) {
          ((__bf16*)Cv)[(size_t)row * N + col] = f2bf(acc[mi][ni][r] + bv);
        } else {
          ((float*)Cv)[(size_t)row * N + col] = acc[mi][ni][r];
        }
      }
    }
}

// ---------------------------------------------------------------------------
// Kernel 2: RoPE on q and k, packed to head-major bf16
// ---------------------------------------------------------------------------
__global__ __launch_bounds__(256) void rope_pack_kernel(
    const __bf16* __restrict__ qkv, const float* __restrict__ rope,
    __bf16* __restrict__ Qo, __bf16* __restrict__ Ko) {
  int idx = blockIdx.x * 256 + threadIdx.x;
  int p = idx & 63;
  int hh = (idx >> 6) % (NH_ + NG_);
  int s = ((idx >> 6) / (NH_ + NG_)) % S_;
  int b = idx / (64 * (NH_ + NG_) * S_);
  if (b >= B_) return;
  const __bf16* row = qkv + (size_t)(b * S_ + s) * NQKV;
  float c = rope[((size_t)s * 64 + p) * 2 + 0];
  float sn = rope[((size_t)s * 64 + p) * 2 + 1];
  float x0, x1;
  __bf16* dst;
  if (hh < NH_) {
    x0 = bf2f(row[hh * HD_ + 2 * p]);
    x1 = bf2f(row[hh * HD_ + 2 * p + 1]);
    dst = Qo + ((size_t)(b * NH_ + hh) * S_ + s) * HD_ + 2 * p;
  } else {
    int g = hh - NH_;
    x0 = bf2f(row[H_ + g * HD_ + 2 * p]);
    x1 = bf2f(row[H_ + g * HD_ + 2 * p + 1]);
    dst = Ko + ((size_t)(b * NG_ + g) * S_ + s) * HD_ + 2 * p;
  }
  dst[0] = f2bf(x0 * c - x1 * sn);
  dst[1] = f2bf(x1 * c + x0 * sn);
}

// ---------------------------------------------------------------------------
// Kernel 3: V packed transposed bf16: Vt [B,NG,128,S]
// ---------------------------------------------------------------------------
__global__ __launch_bounds__(256) void vpack_kernel(
    const __bf16* __restrict__ qkv, __bf16* __restrict__ Vt) {
  int idx = blockIdx.x * 256 + threadIdx.x;
  int s = idx % S_;
  int d = (idx / S_) % HD_;
  int g = (idx / (S_ * HD_)) % NG_;
  int b = idx / (S_ * HD_ * NG_);
  if (b >= B_) return;
  Vt[((size_t)(b * NG_ + g) * HD_ + d) * S_ + s] =
      qkv[(size_t)(b * S_ + s) * NQKV + H_ + NG_ * HD_ + g * HD_ + d];
}

// ---------------------------------------------------------------------------
// Kernel 4: causal GQA attention, flash-style online softmax.
// ---------------------------------------------------------------------------
__global__ __launch_bounds__(256) void attn_kernel(
    const __bf16* __restrict__ Q, const __bf16* __restrict__ Kc,
    const __bf16* __restrict__ Vt, __bf16* __restrict__ AO) {
  const int QB = S_ / 128;  // 16
  int blk = blockIdx.x;
  int qblk = blk % QB;
  int h = (blk / QB) % NH_;
  int b = blk / (QB * NH_);
  int g = h / (NH_ / NG_);
  int tid = threadIdx.x;
  int lane = tid & 31, wave = tid >> 5;
  int half = lane >> 4, l16 = lane & 15;
  int qr0 = qblk * 128 + wave * 16;

  const __bf16* Qp = Q + ((size_t)(b * NH_ + h) * S_ + qr0) * HD_;
  const __bf16* Kp = Kc + (size_t)(b * NG_ + g) * S_ * HD_;
  const __bf16* Vp = Vt + (size_t)(b * NG_ + g) * HD_ * S_;

  v16bf qf[4];
  {
    const __bf16* qrow = Qp + (size_t)l16 * HD_;
#pragma unroll
    for (int ds = 0; ds < 4; ++ds) {
      const __bf16* p = qrow + ds * 32 + half * 8;
      qf[ds] = pack16(*(const v8bf*)p, *(const v8bf*)(p + 16));
    }
  }

  v8f O[8];
#pragma unroll
  for (int dt = 0; dt < 8; ++dt) O[dt] = zero8();
  float mrow[8], lrow[8];
#pragma unroll
  for (int r = 0; r < 8; ++r) { mrow[r] = -3.0e38f; lrow[r] = 0.f; }

  __shared__ __bf16 Pl[8][16][40];  // per-wave P tile

  const float scale = 0.088388347648318447f;  // 1/sqrt(128)
  const int kend = qr0 + 16;                  // causal limit (wave-uniform)
  for (int kb = 0; kb < kend; kb += 32) {
    float sv[2][8];
#pragma unroll
    for (int ks = 0; ks < 2; ++ks) {
      v8f s = zero8();
      const __bf16* krow = Kp + (size_t)(kb + ks * 16 + l16) * HD_;
#pragma unroll
      for (int ds = 0; ds < 4; ++ds) {
        const __bf16* p = krow + ds * 32 + half * 8;
        v16bf kf = pack16(*(const v8bf*)p, *(const v8bf*)(p + 16));
        s = __builtin_amdgcn_wmma_f32_16x16x32_bf16(
            false, qf[ds], false, kf, (short)0, s, false, false);
      }
      int col = kb + ks * 16 + l16;
#pragma unroll
      for (int r = 0; r < 8; ++r) {
        int row = qr0 + r + half * 8;
        float v = s[r] * scale;
        sv[ks][r] = (col > row) ? -3.0e38f : v;
      }
    }
#pragma unroll
    for (int r = 0; r < 8; ++r) {
      float mx = fmaxf(sv[0][r], sv[1][r]);
#pragma unroll
      for (int off = 1; off < 16; off <<= 1)
        mx = fmaxf(mx, __shfl_xor(mx, off, 16));
      float mnew = fmaxf(mrow[r], mx);
      float corr = __expf(mrow[r] - mnew);
      float p0 = __expf(sv[0][r] - mnew);
      float p1 = __expf(sv[1][r] - mnew);
      float sum = p0 + p1;
#pragma unroll
      for (int off = 1; off < 16; off <<= 1)
        sum += __shfl_xor(sum, off, 16);
      lrow[r] = lrow[r] * corr + sum;
      mrow[r] = mnew;
#pragma unroll
      for (int dt = 0; dt < 8; ++dt) O[dt][r] *= corr;
      Pl[wave][r + half * 8][l16] = f2bf(p0);
      Pl[wave][r + half * 8][16 + l16] = f2bf(p1);
    }
    const __bf16* prow = &Pl[wave][l16][half * 8];
    v16bf pf = pack16(*(const v8bf*)prow, *(const v8bf*)(prow + 16));
#pragma unroll
    for (int dt = 0; dt < 8; ++dt) {
      const __bf16* vrow = Vp + (size_t)(dt * 16 + l16) * S_ + kb + half * 8;
      v16bf vf = pack16(*(const v8bf*)vrow, *(const v8bf*)(vrow + 16));
      O[dt] = __builtin_amdgcn_wmma_f32_16x16x32_bf16(
          false, pf, false, vf, (short)0, O[dt], false, false);
    }
  }
#pragma unroll
  for (int r = 0; r < 8; ++r) {
    float inv = 1.0f / lrow[r];
    size_t row = (size_t)b * S_ + qr0 + r + half * 8;
#pragma unroll
    for (int dt = 0; dt < 8; ++dt) {
      int col = h * HD_ + dt * 16 + l16;
      AO[row * H_ + col] = f2bf(O[dt][r] * inv);
    }
  }
}

// ---------------------------------------------------------------------------
extern "C" void kernel_launch(void* const* d_in, const int* in_sizes, int n_in,
                              void* d_out, int out_size, void* d_ws,
                              size_t ws_size, hipStream_t stream) {
  (void)in_sizes; (void)n_in; (void)out_size; (void)ws_size;
  const float* x    = (const float*)d_in[0];
  const float* rope = (const float*)d_in[1];
  const float* Wqkv = (const float*)d_in[2];
  const float* bqkv = (const float*)d_in[3];
  const float* Wd   = (const float*)d_in[4];
  float* out = (float*)d_out;

  char* ws = (char*)d_ws;
  size_t off = 0;
  __bf16* xbf   = (__bf16*)(ws + off); off += (size_t)MROWS * H_ * 2;        // 33.5 MB
  __bf16* AO    = xbf;  // aliased: xbf dead before attention writes AO
  __bf16* WqkvT = (__bf16*)(ws + off); off += (size_t)NQKV * H_ * 2;         // 50.3 MB
  __bf16* WdT   = (__bf16*)(ws + off); off += (size_t)H_ * H_ * 2;           // 33.5 MB
  __bf16* qkv   = (__bf16*)(ws + off); off += (size_t)MROWS * NQKV * 2;      // 50.3 MB
  __bf16* Qb    = (__bf16*)(ws + off); off += (size_t)B_ * NH_ * S_ * HD_ * 2;
  __bf16* Kb    = (__bf16*)(ws + off); off += (size_t)B_ * NG_ * S_ * HD_ * 2;
  __bf16* Vt    = (__bf16*)(ws + off); off += (size_t)B_ * NG_ * HD_ * S_ * 2;

  cvt_bf16_kernel<<<(MROWS * H_) / 256, 256, 0, stream>>>(x, xbf, MROWS * H_);
  transpose_cvt_kernel<<<dim3(H_ / 32, NQKV / 32), 256, 0, stream>>>(
      Wqkv, WqkvT, H_, NQKV);
  transpose_cvt_kernel<<<dim3(H_ / 32, H_ / 32), 256, 0, stream>>>(
      Wd, WdT, H_, H_);

  gemm_kernel<true><<<dim3(NQKV / 128, MROWS / 128), 256, 0, stream>>>(
      xbf, WqkvT, bqkv, qkv, NQKV);

  rope_pack_kernel<<<(B_ * S_ * (NH_ + NG_) * 64) / 256, 256, 0, stream>>>(
      qkv, rope, Qb, Kb);
  vpack_kernel<<<(B_ * NG_ * HD_ * S_) / 256, 256, 0, stream>>>(qkv, Vt);

  attn_kernel<<<B_ * NH_ * (S_ / 128), 256, 0, stream>>>(Qb, Kb, Vt, AO);

  gemm_kernel<false><<<dim3(H_ / 128, MROWS / 128), 256, 0, stream>>>(
      AO, WdT, nullptr, out, H_);
}
